// FastMLayer_85710367359287
// MI455X (gfx1250) — matrix operand order repaired and checked by reading, not compiled, not run on hardware
//
#include <hip/hip_runtime.h>
#include <hip/hip_bf16.h>

typedef __bf16 bf16;
typedef __attribute__((ext_vector_type(16))) __bf16 v16bf;
typedef __attribute__((ext_vector_type(8)))  __bf16 v8bf;
typedef __attribute__((ext_vector_type(8)))  float  v8f;

#define RMS_EPS 1.1920928955078125e-07f

// ---------------------------------------------------------------------------
// Transpose + f32 -> bf16 convert:  in[K][N] (row-major) -> out[N][K] bf16
// ---------------------------------------------------------------------------
__global__ __launch_bounds__(256) void transpose_to_bf16(
    const float* __restrict__ in, bf16* __restrict__ out, int K, int N) {
  __shared__ float tile[32][33];
  int nb = blockIdx.x * 32;
  int kb = blockIdx.y * 32;
  int tx = threadIdx.x;          // 0..31
  int ty = threadIdx.y;          // 0..7
#pragma unroll
  for (int i = 0; i < 32; i += 8)
    tile[ty + i][tx] = in[(size_t)(kb + ty + i) * N + (nb + tx)];
  __syncthreads();
#pragma unroll
  for (int i = 0; i < 32; i += 8)
    out[(size_t)(nb + ty + i) * K + (kb + tx)] = (bf16)tile[tx][ty + i];
}

// ---------------------------------------------------------------------------
// RMS norm (no scale) + convert to bf16. One 256-thread block per row.
// ---------------------------------------------------------------------------
__global__ __launch_bounds__(256) void rmsnorm_to_bf16(
    const float* __restrict__ x, bf16* __restrict__ xn, int D) {
  __shared__ float red[256];
  int row = blockIdx.x;
  const float* xr = x + (size_t)row * D;
  bf16* orow = xn + (size_t)row * D;
  float ss = 0.f;
  for (int i = threadIdx.x * 4; i < D; i += 256 * 4) {
    float4 v = *(const float4*)(xr + i);
    ss += v.x * v.x + v.y * v.y + v.z * v.z + v.w * v.w;
  }
  red[threadIdx.x] = ss;
  __syncthreads();
  for (int off = 128; off > 0; off >>= 1) {
    if (threadIdx.x < off) red[threadIdx.x] += red[threadIdx.x + off];
    __syncthreads();
  }
  float inv = rsqrtf(red[0] / (float)D + RMS_EPS);
  for (int i = threadIdx.x * 4; i < D; i += 256 * 4) {
    float4 v = *(const float4*)(xr + i);
    bf16 o0 = (bf16)(v.x * inv);
    bf16 o1 = (bf16)(v.y * inv);
    bf16 o2 = (bf16)(v.z * inv);
    bf16 o3 = (bf16)(v.w * inv);
    orow[i + 0] = o0; orow[i + 1] = o1; orow[i + 2] = o2; orow[i + 3] = o3;
  }
}

// ---------------------------------------------------------------------------
// Fragment loaders matching CDNA5 16-bit WMMA VGPR layouts (ISA 7.12.2).
// A: row-major [M,K].  lanes 0-15: M=lane, elems j0-7 -> K=k0..k0+7,
//    j8-15 -> K=k0+16..k0+23 ; lanes 16-31: M=lane-16, K=k0+8.. / k0+24..
// ---------------------------------------------------------------------------
__device__ inline v16bf load_a_frag(const bf16* __restrict__ A, int lda,
                                    int row0, int k0, int lane) {
  int r = row0 + (lane & 15);
  int kbase = k0 + ((lane >> 4) << 3);
  const bf16* p = A + (size_t)r * lda + kbase;
  v16bf a;
  *(v8bf*)&a           = *(const v8bf*)(p);
  *(((v8bf*)&a) + 1)   = *(const v8bf*)(p + 16);
  return a;
}

// B: from transposed weights Bt[N][K] (row-major, contiguous K).
// lanes 0-15: N=lane, elems j0-7 -> K=k0..k0+7, j8-15 -> K=k0+16..k0+23 ;
// lanes 16-31: N=lane-16, K=k0+8.. / k0+24..
__device__ inline v16bf load_b_frag(const bf16* __restrict__ Bt, int ldb,
                                    int n0, int k0, int lane) {
  int n = n0 + (lane & 15);
  int kbase = k0 + ((lane >> 4) << 3);
  const bf16* p = Bt + (size_t)n * ldb + kbase;
  v16bf b;
  *(v8bf*)&b           = *(const v8bf*)(p);
  *(((v8bf*)&b) + 1)   = *(const v8bf*)(p + 16);
  return b;
}

// ---------------------------------------------------------------------------
// Wave-tiled bf16 GEMM: C[M,N] = A[M,K] * Bt[N,K]^T + bias.
// Each wave computes a 64x32 tile (WM=4 x WN=2 of 16x16 WMMA tiles).
// MODE 0: store bf16. MODE 1: store f32 with strict-upper 16x16 mask.
// MODE 2: store f32.
// ---------------------------------------------------------------------------
template <int MODE>
__global__ __launch_bounds__(128, 1) void gemm_wmma_bf16(
    const bf16* __restrict__ A, const bf16* __restrict__ Bt,
    const float* __restrict__ bias, void* __restrict__ out,
    int M, int N, int K) {
  constexpr int WM = 4, WN = 2;
  int lane = threadIdx.x & 31;
  int wave = threadIdx.x >> 5;
  int tilesN = N / (WN * 16);
  int w = blockIdx.x * 4 + wave;
  int tm = w / tilesN;
  int tn = w % tilesN;
  int m0 = tm * WM * 16;
  int n0 = tn * WN * 16;

  int col = lane & 15;
  int hi = lane >> 4;

  v8f acc[WM][WN];
#pragma unroll
  for (int wn = 0; wn < WN; ++wn) {
    float bv = bias[n0 + wn * 16 + col];
#pragma unroll
    for (int wm = 0; wm < WM; ++wm) {
#pragma unroll
      for (int r = 0; r < 8; ++r) acc[wm][wn][r] = bv;
    }
  }

  for (int k0 = 0; k0 < K; k0 += 32) {
    v16bf a[WM], b[WN];
#pragma unroll
    for (int wm = 0; wm < WM; ++wm)
      a[wm] = load_a_frag(A, K, m0 + wm * 16, k0, lane);
#pragma unroll
    for (int wn = 0; wn < WN; ++wn)
      b[wn] = load_b_frag(Bt, K, n0 + wn * 16, k0, lane);
#pragma unroll
    for (int wm = 0; wm < WM; ++wm)
#pragma unroll
      for (int wn = 0; wn < WN; ++wn)
        acc[wm][wn] = __builtin_amdgcn_wmma_f32_16x16x32_bf16(
            false, a[wm], false, b[wn], (short)0, acc[wm][wn], false, false);
  }

#pragma unroll
  for (int wm = 0; wm < WM; ++wm) {
#pragma unroll
    for (int wn = 0; wn < WN; ++wn) {
#pragma unroll
      for (int r = 0; r < 8; ++r) {
        int row = m0 + wm * 16 + r + 8 * hi;
        int c = n0 + wn * 16 + col;
        float v = acc[wm][wn][r];
        if (MODE == 0) {
          ((bf16*)out)[(size_t)row * N + c] = (bf16)v;
        } else if (MODE == 1) {
          int ti = (c >> 4) & 15, tj = c & 15;  // 16x16 position within 256
          ((float*)out)[(size_t)row * N + c] = (tj > ti) ? v : 0.f;
        } else {
          ((float*)out)[(size_t)row * N + c] = v;
        }
      }
    }
  }
}

// ---------------------------------------------------------------------------
// Per-row 16x16 matrix exponential via Taylor (exact: M strictly upper =>
// nilpotent, 15 terms suffice). One wave per row. term lives in LDS as bf16
// (A-layout source), M is held as a fixed B fragment (K padded to 32 with
// zeros), res accumulated in f32 D-layout registers. Fully unrolled so the
// 1/k factors become inline literal multiplies (no runtime divides).
// ---------------------------------------------------------------------------
__global__ __launch_bounds__(128, 1) void expm_taylor_wmma(
    const float* __restrict__ flatM, bf16* __restrict__ flatExp) {
  __shared__ __align__(16) bf16 termLDS[4][256];
  int lane = threadIdx.x & 31;
  int wave = threadIdx.x >> 5;
  int row = blockIdx.x * 4 + wave;

  const float* Mp = flatM + (size_t)row * 256;
  int col = lane & 15;
  int hi = lane >> 4;
  bool lo = (hi == 0);

  // B fragment of M: elems j0-7 hold K=8*hi+j (real), elems j8-15 hold
  // K=16..31 which are zero-padded.
  v16bf bfrag;
#pragma unroll
  for (int j = 0; j < 16; ++j) {
    float v = (j < 8) ? Mp[(j + 8 * hi) * 16 + col] : 0.f;
    bfrag[j] = (bf16)v;
  }

  // term = I (bf16 in LDS, row-major 16x16)
  bf16* T = termLDS[wave];
#pragma unroll
  for (int e = 0; e < 8; ++e) {
    int idx = e * 32 + lane;
    int r = idx >> 4, c = idx & 15;
    T[idx] = (bf16)((r == c) ? 1.f : 0.f);
  }

  // res = I in D-layout registers
  v8f res;
#pragma unroll
  for (int r = 0; r < 8; ++r) res[r] = ((r + 8 * hi) == col) ? 1.f : 0.f;

  int ar = lane & 15;
  int kb = hi << 3;

#pragma unroll
  for (int k = 1; k <= 15; ++k) {
    // A fragment of term from LDS; K=16..31 zero in registers.
    v8bf lowv = *(const v8bf*)(T + ar * 16 + kb);
    v16bf afrag;
#pragma unroll
    for (int j = 0; j < 8; ++j) afrag[j] = lowv[j];
#pragma unroll
    for (int j = 8; j < 16; ++j) afrag[j] = (bf16)0.f;

    v8f d;
#pragma unroll
    for (int r = 0; r < 8; ++r) d[r] = 0.f;
    d = __builtin_amdgcn_wmma_f32_16x16x32_bf16(
        false, afrag, false, bfrag, (short)0, d, false, false);

    const float s = 1.f / (float)k;  // compile-time constant after unroll
#pragma unroll
    for (int r = 0; r < 8; ++r) {
      float t = d[r] * s;
      res[r] += t;
      d[r] = t;
    }
    // write term back (D-layout -> row-major LDS); skip after last term
    if (k < 15) {
#pragma unroll
      for (int r = 0; r < 8; ++r)
        T[(r + 8 * hi) * 16 + col] = (bf16)d[r];
    }
  }

  bf16* O = flatExp + (size_t)row * 256;
#pragma unroll
  for (int r = 0; r < 8; ++r)
    O[(r + 8 * hi) * 16 + col] = (bf16)res[r];
}

// ---------------------------------------------------------------------------
// Launch
// ---------------------------------------------------------------------------
extern "C" void kernel_launch(void* const* d_in, const int* in_sizes, int n_in,
                              void* d_out, int out_size, void* d_ws, size_t ws_size,
                              hipStream_t stream) {
  const int NROWS = 8192, IN = 4096, EMB = 1024, NN2 = 256, OUT = 4096;

  const float* x   = (const float*)d_in[0];
  const float* W_u = (const float*)d_in[1];
  const float* b_u = (const float*)d_in[2];
  const float* W_t = (const float*)d_in[3];
  const float* b_t = (const float*)d_in[4];
  const float* W_s = (const float*)d_in[5];
  const float* b_s = (const float*)d_in[6];
  float* out = (float*)d_out;

  size_t off = 0;
  auto carve = [&](size_t bytes) {
    void* p = (char*)d_ws + off;
    off += (bytes + 255) & ~(size_t)255;
    return p;
  };
  bf16*  xn_bf   = (bf16*)carve((size_t)NROWS * IN * 2);
  bf16*  Wu_t    = (bf16*)carve((size_t)EMB * IN * 2);     // [EMB][IN]
  bf16*  Wt_t    = (bf16*)carve((size_t)NN2 * EMB * 2);    // [256][EMB]
  bf16*  Ws_t    = (bf16*)carve((size_t)OUT * NN2 * 2);    // [OUT][256]
  bf16*  lat_bf  = (bf16*)carve((size_t)NROWS * EMB * 2);
  float* flat_m  = (float*)carve((size_t)NROWS * NN2 * 4);
  bf16*  fexp_bf = (bf16*)carve((size_t)NROWS * NN2 * 2);

  dim3 tb(32, 8);
  // W_u: [IN, EMB] -> Wu_t [EMB, IN]
  transpose_to_bf16<<<dim3(EMB / 32, IN / 32), tb, 0, stream>>>(W_u, Wu_t, IN, EMB);
  // W_t: [EMB, 256] -> Wt_t [256, EMB]
  transpose_to_bf16<<<dim3(NN2 / 32, EMB / 32), tb, 0, stream>>>(W_t, Wt_t, EMB, NN2);
  // W_s: [256, OUT] -> Ws_t [OUT, 256]
  transpose_to_bf16<<<dim3(OUT / 32, NN2 / 32), tb, 0, stream>>>(W_s, Ws_t, NN2, OUT);

  rmsnorm_to_bf16<<<NROWS, 256, 0, stream>>>(x, xn_bf, IN);

  // GEMM1: [8192,4096] x [4096,1024] -> latent bf16
  {
    int waves = (NROWS / 64) * (EMB / 32);
    gemm_wmma_bf16<0><<<waves / 4, 128, 0, stream>>>(
        xn_bf, Wu_t, b_u, (void*)lat_bf, NROWS, EMB, IN);
  }
  // GEMM2: [8192,1024] x [1024,256] -> flat_m f32 (strict-upper masked)
  {
    int waves = (NROWS / 64) * (NN2 / 32);
    gemm_wmma_bf16<1><<<waves / 4, 128, 0, stream>>>(
        lat_bf, Wt_t, b_t, (void*)flat_m, NROWS, NN2, EMB);
  }
  // expm per row
  expm_taylor_wmma<<<NROWS / 4, 128, 0, stream>>>(flat_m, fexp_bf);

  // GEMM3: [8192,256] x [256,4096] -> out f32
  {
    int waves = (NROWS / 64) * (OUT / 32);
    gemm_wmma_bf16<2><<<waves / 4, 128, 0, stream>>>(
        fexp_bf, Ws_t, b_s, (void*)out, NROWS, OUT, NN2);
  }
}
